// GEncoder_25984552141076
// MI455X (gfx1250) — matrix operand order repaired and checked by reading, not compiled, not run on hardware
//
#include <hip/hip_runtime.h>
#include <stdint.h>
#include <stddef.h>

// ---------- types ----------
typedef __attribute__((ext_vector_type(16))) __bf16  v16bf;
typedef __attribute__((ext_vector_type(8)))  float   v8f;
typedef __attribute__((ext_vector_type(4)))  float   f32x4;
typedef __attribute__((ext_vector_type(4)))  unsigned u32x4;

// pack two f32 -> bf16x2 dword: round-to-nearest (ties away), then v_perm_b32
// merges the two high halves in a single instruction.  3 VALU ops total.
static __device__ __forceinline__ unsigned pack_bf16x2(float lo, float hi) {
  unsigned a = __float_as_uint(lo) + 0x8000u;
  unsigned b = __float_as_uint(hi) + 0x8000u;
  // result = { b[31:16], a[31:16] } : sel 3,2 -> S1(=b) bytes 3,2 ; sel 7,6 -> S0(=a) bytes 3,2
  return __builtin_amdgcn_perm(a, b, 0x03020706u);
}

// ---------- pack W (Kdim x 128, row-major f32) into B-fragment bf16 layout ----------
// dword[(g*128 + n)*8 + j] = pack( W[(g*16+2j)*128+n], W[(g*16+2j+1)*128+n] )
// so a lane's B fragment is 2 x b128: j = 4*khalf..4*khalf+3 of K-pair groups g and g+1 (same n).
__global__ void pack_w_kernel(const float* __restrict__ W, unsigned* __restrict__ bpk) {
  int t = blockIdx.x * blockDim.x + threadIdx.x;   // t < Kdim*64
  int n = t & 127;
  int j = (t >> 7) & 7;
  int g = t >> 10;
  float lo = W[(size_t)(g * 16 + 2 * j)     * 128 + n];
  float hi = W[(size_t)(g * 16 + 2 * j + 1) * 128 + n];
  bpk[((size_t)g * 128 + n) * 8 + j] = pack_bf16x2(lo, hi);
}

// ---------- h = x @ W via v_wmma_f32_16x16x32_bf16 ----------
// block = 64 threads (2 waves). wave w -> 32 rows [blk*64 + 32w, +32), all 128 cols.
// Per k-step of 32: 2 A fragments (8 x b128 NT loads), 8 B fragments (16 x b128,
// L0/L2-hot), 16 WMMAs.  B is loaded once and feeds both row-halves.
__global__ void __launch_bounds__(64) gemm_bf16_kernel(const float* __restrict__ x,
                                                       const unsigned* __restrict__ bpk,
                                                       float* __restrict__ h,
                                                       int Kdim) {
  const int lane  = threadIdx.x & 31;
  const int wave  = threadIdx.x >> 5;
  const int l15   = lane & 15;
  const int khalf = lane >> 4;                    // 0: K 0..7/16..23, 1: K 8..15/24..31
  const int mb    = blockIdx.x * 64 + wave * 32;

  const float* rowp0 = x + (size_t)(mb + l15) * Kdim + khalf * 8;        // rows mb..mb+15
  const float* rowp1 = rowp0 + (size_t)16 * Kdim;                        // rows mb+16..mb+31
  const u32x4* bq = (const u32x4*)bpk;

  v8f acc0[8], acc1[8];
#pragma unroll
  for (int i = 0; i < 8; ++i) { acc0[i] = (v8f)0.0f; acc1[i] = (v8f)0.0f; }

  for (int kb = 0; kb < Kdim; kb += 32) {
    // ---- A fragments: 16x32 bf16 each (ISA layout: lane holds its row;
    //      VGPR0-3 = K kb+khalf*8+0..7, VGPR4-7 = same +16) ----
    const f32x4* ap0 = (const f32x4*)(rowp0 + kb);
    const f32x4* ap1 = (const f32x4*)(rowp1 + kb);
    f32x4 a0 = __builtin_nontemporal_load(ap0);
    f32x4 a1 = __builtin_nontemporal_load(ap0 + 1);
    f32x4 a2 = __builtin_nontemporal_load(ap0 + 4);
    f32x4 a3 = __builtin_nontemporal_load(ap0 + 5);
    f32x4 c0 = __builtin_nontemporal_load(ap1);
    f32x4 c1 = __builtin_nontemporal_load(ap1 + 1);
    f32x4 c2 = __builtin_nontemporal_load(ap1 + 4);
    f32x4 c3 = __builtin_nontemporal_load(ap1 + 5);
    __builtin_prefetch((const void*)(rowp0 + kb + 256), 0, 1);  // global_prefetch_b8
    __builtin_prefetch((const void*)(rowp1 + kb + 256), 0, 1);

    union { v16bf v; unsigned u[8]; } A0, A1;
    A0.u[0] = pack_bf16x2(a0.x, a0.y);  A0.u[1] = pack_bf16x2(a0.z, a0.w);
    A0.u[2] = pack_bf16x2(a1.x, a1.y);  A0.u[3] = pack_bf16x2(a1.z, a1.w);
    A0.u[4] = pack_bf16x2(a2.x, a2.y);  A0.u[5] = pack_bf16x2(a2.z, a2.w);
    A0.u[6] = pack_bf16x2(a3.x, a3.y);  A0.u[7] = pack_bf16x2(a3.z, a3.w);
    A1.u[0] = pack_bf16x2(c0.x, c0.y);  A1.u[1] = pack_bf16x2(c0.z, c0.w);
    A1.u[2] = pack_bf16x2(c1.x, c1.y);  A1.u[3] = pack_bf16x2(c1.z, c1.w);
    A1.u[4] = pack_bf16x2(c2.x, c2.y);  A1.u[5] = pack_bf16x2(c2.z, c2.w);
    A1.u[6] = pack_bf16x2(c3.x, c3.y);  A1.u[7] = pack_bf16x2(c3.z, c3.w);

    const int g0 = (kb >> 4);                           // lane's K-pair group
    const size_t bbase = ((size_t)g0 * 128 + l15) * 2 + khalf;  // uint4 index

#pragma unroll
    for (int nb = 0; nb < 8; ++nb) {
      union { v16bf v; u32x4 q[2]; } B;
      B.q[0] = bq[bbase + (size_t)nb * 32];
      B.q[1] = bq[bbase + (size_t)nb * 32 + 256];
      acc0[nb] = __builtin_amdgcn_wmma_f32_16x16x32_bf16(
          false, A0.v, false, B.v, (short)0, acc0[nb], false, false);
      acc1[nb] = __builtin_amdgcn_wmma_f32_16x16x32_bf16(
          false, A1.v, false, B.v, (short)0, acc1[nb], false, false);
    }
  }

  // C/D layout: lane 0-15 -> N=lane, M=vgpr; lane 16-31 -> N=lane-16, M=vgpr+8
  float* hp0 = h + (size_t)(mb + khalf * 8) * 128 + l15;
  float* hp1 = hp0 + (size_t)16 * 128;
#pragma unroll
  for (int nb = 0; nb < 8; ++nb)
#pragma unroll
    for (int r = 0; r < 8; ++r) {
      hp0[(size_t)r * 128 + nb * 16] = acc0[nb][r];
      hp1[(size_t)r * 128 + nb * 16] = acc1[nb][r];
    }
}

// ---------- sparse side ----------
__global__ void init_deg_kernel(float* __restrict__ deg, int N) {
  int i = blockIdx.x * blockDim.x + threadIdx.x;
  if (i < N) deg[i] = 1.0f;   // self-loop weight
}

__global__ void ew_deg_kernel(const float* __restrict__ x, const int* __restrict__ edge,
                              float* __restrict__ ew, float* __restrict__ deg,
                              int E, int N) {
  int e = blockIdx.x * blockDim.x + threadIdx.x;
  if (e >= E) return;
  int r = edge[e];
  int c = edge[E + e];
  float w = x[(size_t)r * N + c];
  ew[e] = w;
  atomicAdd(&deg[c], w);
}

__global__ void dinv_kernel(float* __restrict__ deg, int N) {
  int i = blockIdx.x * blockDim.x + threadIdx.x;
  if (i >= N) return;
  float d = deg[i];
  deg[i] = (d > 0.0f) ? rsqrtf(fmaxf(d, 1e-30f)) : 0.0f;
}

// out = b + dinv[i]^2 * h[i]   (bias + self-loop message)
__global__ void out_init_kernel(const float* __restrict__ h, const float* __restrict__ dinv,
                                const float* __restrict__ b, float* __restrict__ out, int N) {
  int t = blockIdx.x * blockDim.x + threadIdx.x;
  if (t >= N * 128) return;
  int i = t >> 7;
  int d = t & 127;
  float s = dinv[i];
  out[t] = b[d] + s * s * h[t];
}

// out[col] += dinv[row]*ew*dinv[col] * h[row]
__global__ void scatter_kernel(const float* __restrict__ h, const float* __restrict__ dinv,
                               const float* __restrict__ ew, const int* __restrict__ edge,
                               float* __restrict__ out, int E) {
  size_t t = (size_t)blockIdx.x * blockDim.x + threadIdx.x;
  size_t e = t >> 7;
  int d = (int)(t & 127);
  if (e >= (size_t)E) return;
  int r = edge[e];
  int c = edge[E + e];
  float norm = dinv[r] * ew[e] * dinv[c];
  atomicAdd(&out[(size_t)c * 128 + d], norm * h[(size_t)r * 128 + d]);
}

__global__ void relu_kernel(float* __restrict__ out, size_t n) {
  size_t t = (size_t)blockIdx.x * blockDim.x + threadIdx.x;
  if (t < n) out[t] = fmaxf(out[t], 0.0f);
}

// ---------- host-side orchestration ----------
static void run_gcn(const float* x, const int* edge, const float* W, const float* b,
                    int N, int E,
                    float* h, unsigned* bpk, float* deg, float* ew, float* out,
                    hipStream_t stream) {
  // pack W into bf16 B-fragment layout (N == Kdim)
  pack_w_kernel<<<(N * 64) / 256, 256, 0, stream>>>(W, bpk);
  init_deg_kernel<<<(N + 255) / 256, 256, 0, stream>>>(deg, N);
  ew_deg_kernel<<<(E + 255) / 256, 256, 0, stream>>>(x, edge, ew, deg, E, N);
  dinv_kernel<<<(N + 255) / 256, 256, 0, stream>>>(deg, N);
  gemm_bf16_kernel<<<N / 64, 64, 0, stream>>>(x, bpk, h, N);
  out_init_kernel<<<(N * 128) / 256, 256, 0, stream>>>(h, deg, b, out, N);
  scatter_kernel<<<((size_t)E * 128) / 256, 256, 0, stream>>>(h, deg, ew, edge, out, E);
}

extern "C" void kernel_launch(void* const* d_in, const int* in_sizes, int n_in,
                              void* d_out, int out_size, void* d_ws, size_t ws_size,
                              hipStream_t stream) {
  (void)in_sizes; (void)n_in; (void)out_size; (void)ws_size;
  constexpr int NC = 8192, ND = 4096;
  constexpr int EC = 262144, ED = 131072;
  constexpr int D = 128;

  const float* cell_feat = (const float*)d_in[0];
  const int*   cell_edge = (const int*)d_in[1];
  const float* W_cell    = (const float*)d_in[2];
  const float* b_cell    = (const float*)d_in[3];
  const float* drug_feat = (const float*)d_in[4];
  const int*   drug_edge = (const int*)d_in[5];
  const float* W_drug    = (const float*)d_in[6];
  const float* b_drug    = (const float*)d_in[7];

  float* out_cell = (float*)d_out;
  float* out_drug = out_cell + (size_t)NC * D;

  // workspace carve-up (all offsets 256B aligned)
  char* ws = (char*)d_ws;
  size_t off = 0;
  float*    h_cell  = (float*)(ws + off);  off += (size_t)NC * D * 4;    // 4 MB
  float*    h_drug  = (float*)(ws + off);  off += (size_t)ND * D * 4;    // 2 MB
  unsigned* bpk_cell= (unsigned*)(ws + off); off += (size_t)NC * 64 * 4; // 2 MB
  unsigned* bpk_drug= (unsigned*)(ws + off); off += (size_t)ND * 64 * 4; // 1 MB
  float*    deg_cell= (float*)(ws + off);  off += (size_t)NC * 4;
  float*    deg_drug= (float*)(ws + off);  off += (size_t)ND * 4;
  float*    ew_cell = (float*)(ws + off);  off += (size_t)EC * 4;
  float*    ew_drug = (float*)(ws + off);  off += (size_t)ED * 4;

  run_gcn(cell_feat, cell_edge, W_cell, b_cell, NC, EC,
          h_cell, bpk_cell, deg_cell, ew_cell, out_cell, stream);
  run_gcn(drug_feat, drug_edge, W_drug, b_drug, ND, ED,
          h_drug, bpk_drug, deg_drug, ew_drug, out_drug, stream);

  size_t total = (size_t)(NC + ND) * D;
  relu_kernel<<<(total + 255) / 256, 256, 0, stream>>>(out_cell, total);
}
